// DepthRefinement_20641612824906
// MI455X (gfx1250) — matrix operand order repaired and checked by reading, not compiled, not run on hardware
//
#include <hip/hip_runtime.h>

// ---------------------------------------------------------------- constants
#define H 481
#define W 641
#define HW (481 * 641)
#define NTAP 81
#define C 128
#define NSUB 4                      // 4 x 16-pixel N-subtiles per wave

typedef __bf16       v16bf __attribute__((ext_vector_type(16)));
typedef float        v8f   __attribute__((ext_vector_type(8)));
typedef unsigned int v4u   __attribute__((ext_vector_type(4)));

union Frag16 {           // 32 bytes: one WMMA bf16 A/B operand per lane
    v16bf h;
    v4u   u[2];
};

__device__ __forceinline__ unsigned short f2bf(float f) {
    unsigned int u = __float_as_uint(f);
    unsigned int r = u + 0x7FFFu + ((u >> 16) & 1u);   // round-to-nearest-even
    return (unsigned short)(r >> 16);
}
__device__ __forceinline__ float bf2f(unsigned short h) {
    return __uint_as_float(((unsigned int)h) << 16);
}

// ---------------------------------------------------------------- stage 1
// One wave per pixel; lanes stride the 81 taps (coalesced 12B/lane reads of
// matrix_c), then a wave32 shuffle reduction.  Memory-bound: ~0.73 GB total
// -> ~31us floor at 23.3 TB/s.
__global__ __launch_bounds__(256)
void stage1_kernel(const float* __restrict__ ma, const float* __restrict__ mc,
                   const float* __restrict__ grid, const float* __restrict__ angle,
                   const unsigned char* __restrict__ valid,
                   float* __restrict__ out_plane) {
    const int lane = threadIdx.x & 31;
    const long pix = (long)blockIdx.x * 8 + (threadIdx.x >> 5);
    if (pix >= (long)HW) return;
    const int y = (int)(pix / W), x = (int)(pix % W);

    const float g0 = grid[pix * 3 + 0];
    const float g1 = grid[pix * 3 + 1];
    const float g2 = grid[pix * 3 + 2];

    float sum_va = 0.0f, sum_eva = 0.0f;
    for (int t = lane; t < NTAP; t += 32) {
        const long base = pix * NTAP + t;
        const float* c = mc + base * 3;
        const float c0 = c[0], c1 = c[1], c2 = c[2];
        const float lm = c0 * g0 + c1 * g1 + c2 * g2;
        const bool cond = lm > 1e-5f;
        const float lower = 1.0f / (cond ? lm : 1.0f);
        float va = valid[base] ? angle[base] : 0.0f;
        va = cond ? va : 0.0f;
        const int i = t / 9, j = t - i * 9;
        const int gy = y + i * 4 - 16, gx = x + j * 4 - 16;   // RATE=4, PAD=16
        float p0 = 0.0f, p1 = 0.0f, p2 = 0.0f;
        if (gy >= 0 && gy < H && gx >= 0 && gx < W) {
            const float* gp = grid + ((long)gy * W + gx) * 3;
            p0 = gp[0]; p1 = gp[1]; p2 = gp[2];
        }
        const float upper = c0 * p0 + c1 * p1 + c2 * p2;
        const float est = lower * upper * ma[base * 3 + 2];
        sum_va  += va;
        sum_eva += est * va;
    }
    for (int off = 16; off > 0; off >>= 1) {
        sum_va  += __shfl_xor(sum_va,  off, 32);
        sum_eva += __shfl_xor(sum_eva, off, 32);
    }
    if (lane == 0) {
        float d = sum_eva / (sum_va + 1e-5f);
        d = fminf(fmaxf(d, 0.0f), 10.0f);
        out_plane[pix] = d;
    }
}

// ---------------------------------------------------------------- layer 1
// 3 input channels, dil=2 pad=2.  Tiny (0.5% of FLOPs) -> plain VALU kernel.
// Writes NHWC bf16 activations for the WMMA pipeline.
__global__ __launch_bounds__(256)
void conv1_kernel(const float* __restrict__ plane, const float* __restrict__ expd,
                  const float* __restrict__ img, const float* __restrict__ w1,
                  const float* __restrict__ b1, unsigned short* __restrict__ act) {
    const long t = (long)blockIdx.x * 256 + threadIdx.x;
    if (t >= (long)HW * C) return;
    const int oc  = (int)(t & (C - 1));
    const long pix = t >> 7;
    const int y = (int)(pix / W), x = (int)(pix % W);

    float acc = b1[oc];
    for (int ky = 0; ky < 3; ++ky) {
        const int sy = y + (ky - 1) * 2;
        if (sy < 0 || sy >= H) continue;
        for (int kx = 0; kx < 3; ++kx) {
            const int sx = x + (kx - 1) * 2;
            if (sx < 0 || sx >= W) continue;
            const long sp = (long)sy * W + sx;
            const float* wr = w1 + ((oc * 3) * 3 + ky) * 3 + kx;  // ic stride 9
            acc += plane[sp] * wr[0]
                 + expd[sp]  * wr[9]
                 + img[sp] * (1.0f / 255.0f) * wr[18];
        }
    }
    act[pix * C + oc] = f2bf(fmaxf(acc, 0.0f));
}

// ---------------------------------------------------------------- weight pack
// f32 (O,I,3,3) -> bf16 [oc][tap][ic] so A fragments are contiguous 16B loads.
__global__ __launch_bounds__(256)
void pack_weights_kernel(const float* __restrict__ w2, const float* __restrict__ w3,
                         const float* __restrict__ w4, const float* __restrict__ w5,
                         const float* __restrict__ w6, unsigned short* __restrict__ wpk) {
    const int per = C * 9 * C;
    const int t = blockIdx.x * 256 + threadIdx.x;
    if (t >= 5 * per) return;
    const int layer = t / per, r = t - layer * per;
    const int oc  = r / (9 * C);
    const int tap = (r / C) % 9;
    const int ic  = r & (C - 1);
    const float* w = (layer == 0) ? w2 : (layer == 1) ? w3
                   : (layer == 2) ? w4 : (layer == 3) ? w5 : w6;
    const int ky = tap / 3, kx = tap - ky * 3;
    wpk[t] = f2bf(w[((oc * C + ic) * 3 + ky) * 3 + kx]);
}

// ---------------------------------------------------------------- WMMA conv
// Implicit GEMM, output-stationary: wave tile = 16 oc x 64 px (4 accumulators),
// block = 8 waves = 128 oc x 64 px.  K = 9 taps x 128 ic = 1152 (36 x K32).
// One A fragment amortized over 4 wmmas -> L2 weight traffic per layer drops
// from ~5.8 GB (16-px tile) to ~1.5 GB, balancing it against WMMA compute.
// Fragment element placement (contiguous k order, shared by A 16x32 and B 32x16):
//   A lane L: M=L%16, K = (L<16?0:16)+i                    -> aoff 0/16
//   B lane L: N=L%16, K = (L<16?0:16)+i                    -> boff 0/16
__global__ __launch_bounds__(256)
void conv_wmma_kernel(const unsigned short* __restrict__ in,
                      unsigned short* __restrict__ out,
                      const unsigned short* __restrict__ wpk,
                      const float* __restrict__ bias, int dil) {
    const int lane = threadIdx.x & 31;
    const int wv   = threadIdx.x >> 5;          // 0..7
    const int n    = lane & 15;                 // pixel column in subtile
    const bool hi  = lane >= 16;
    const int x0   = blockIdx.x * (16 * NSUB);
    const int y    = blockIdx.y;
    const int M    = wv * 16 + n;               // A row (output channel)
    const int aoff = hi ? 16 : 0;
    const int boff = hi ? 16 : 0;

    const v4u zero = {0u, 0u, 0u, 0u};
    v8f acc[NSUB];
#pragma unroll
    for (int s = 0; s < NSUB; ++s)
        acc[s] = (v8f){0.f, 0.f, 0.f, 0.f, 0.f, 0.f, 0.f, 0.f};
    Frag16 a, b;

    for (int tap = 0; tap < 9; ++tap) {
        const int ty = tap / 3, tx = tap - ty * 3;
        const int sy  = y + (ty - 1) * dil;     // pad == dil for every layer
        const bool yok = (sy >= 0) & (sy < H);
        const v4u* abase = reinterpret_cast<const v4u*>(
            wpk + ((M * 9 + tap) * C + aoff));
        if (tap < 8)                            // warm next tap's weight row
            __builtin_prefetch((const void*)(wpk + ((M * 9 + tap + 1) * C)), 0, 0);

        const unsigned short* brow[NSUB];
        bool ok[NSUB];
#pragma unroll
        for (int s = 0; s < NSUB; ++s) {
            const int sx = x0 + s * 16 + n + (tx - 1) * dil;
            ok[s]   = yok & (sx >= 0) & (sx < W);
            brow[s] = in + (((long)sy * W + sx) * C + boff);
        }
#pragma unroll
        for (int cb = 0; cb < 4; ++cb) {        // c0 = 32*cb
            a.u[0] = abase[cb * 4];             // channels c0+aoff    .. +8
            a.u[1] = abase[cb * 4 + 1];         // channels c0+aoff+8  .. +8
#pragma unroll
            for (int s = 0; s < NSUB; ++s) {
                if (ok[s]) {
                    const v4u* bb = reinterpret_cast<const v4u*>(brow[s]);
                    b.u[0] = bb[cb * 4];        // channels c0+boff    .. +8
                    b.u[1] = bb[cb * 4 + 1];    // channels c0+boff+8  .. +8
                } else {
                    b.u[0] = zero;
                    b.u[1] = zero;
                }
                acc[s] = __builtin_amdgcn_wmma_f32_16x16x32_bf16(
                    false, a.h, false, b.h, (short)0, acc[s], false, false);
            }
        }
    }

    // epilogue: bias + relu + bf16 pack, one b128 store per subtile
    const int ocb = wv * 16 + (hi ? 8 : 0);     // C/D: lane holds M = ocb + r
    float bs[8];
#pragma unroll
    for (int r = 0; r < 8; ++r) bs[r] = bias[ocb + r];
#pragma unroll
    for (int s = 0; s < NSUB; ++s) {
        const int xo = x0 + s * 16 + n;
        if (xo < W) {
            unsigned int p[4];
#pragma unroll
            for (int r = 0; r < 4; ++r) {
                float v0 = fmaxf(acc[s][2 * r]     + bs[2 * r],     0.0f);
                float v1 = fmaxf(acc[s][2 * r + 1] + bs[2 * r + 1], 0.0f);
                p[r] = (unsigned int)f2bf(v0) | ((unsigned int)f2bf(v1) << 16);
            }
            v4u pk = {p[0], p[1], p[2], p[3]};
            *reinterpret_cast<v4u*>(out + ((long)y * W + xo) * C + ocb) = pk;
        }
    }
}

// ---------------------------------------------------------------- layer 7
// 128 -> 1 channel, dil=1 pad=1.  Tiny; per-pixel VALU kernel, f32 out.
__global__ __launch_bounds__(256)
void conv7_kernel(const unsigned short* __restrict__ in,
                  const float* __restrict__ w7, const float* __restrict__ b7,
                  float* __restrict__ out) {
    const long pix = (long)blockIdx.x * 256 + threadIdx.x;
    if (pix >= (long)HW) return;
    const int y = (int)(pix / W), x = (int)(pix % W);
    float acc = b7[0];
    for (int tap = 0; tap < 9; ++tap) {
        const int ty = tap / 3, tx = tap - ty * 3;
        const int sy = y + ty - 1, sx = x + tx - 1;
        if (sy < 0 || sy >= H || sx < 0 || sx >= W) continue;
        const unsigned short* row = in + ((long)sy * W + sx) * C;
        const float* wr = w7 + tap;             // w7[ic][tap], ic stride 9
        float s = 0.0f;
#pragma unroll 8
        for (int ic = 0; ic < C; ++ic)
            s += bf2f(row[ic]) * wr[ic * 9];
        acc += s;
    }
    out[pix] = acc;
}

// ---------------------------------------------------------------- launcher
extern "C" void kernel_launch(void* const* d_in, const int* in_sizes, int n_in,
                              void* d_out, int out_size, void* d_ws, size_t ws_size,
                              hipStream_t stream) {
    const float*         ma    = (const float*)d_in[0];
    const float*         mc    = (const float*)d_in[1];
    const float*         grid  = (const float*)d_in[2];
    const float*         angle = (const float*)d_in[3];
    const unsigned char* valid = (const unsigned char*)d_in[4];   // bool array
    const float*         expd  = (const float*)d_in[5];
    const float*         img   = (const float*)d_in[6];
    const float* w1 = (const float*)d_in[7];   const float* b1 = (const float*)d_in[8];
    const float* w2 = (const float*)d_in[9];   const float* b2 = (const float*)d_in[10];
    const float* w3 = (const float*)d_in[11];  const float* b3 = (const float*)d_in[12];
    const float* w4 = (const float*)d_in[13];  const float* b4 = (const float*)d_in[14];
    const float* w5 = (const float*)d_in[15];  const float* b5 = (const float*)d_in[16];
    const float* w6 = (const float*)d_in[17];  const float* b6 = (const float*)d_in[18];
    const float* w7 = (const float*)d_in[19];  const float* b7 = (const float*)d_in[20];
    float* out = (float*)d_out;

    // workspace carve-up (~161 MB): f32 plane + 2 bf16 NHWC act buffers + packed W
    char* ws = (char*)d_ws;
    const size_t planeB = ((size_t)HW * 4 + 255) & ~(size_t)255;
    const size_t actB   = ((size_t)HW * C * 2 + 255) & ~(size_t)255;
    float*          plane = (float*)ws;
    unsigned short* actA  = (unsigned short*)(ws + planeB);
    unsigned short* actBf = (unsigned short*)(ws + planeB + actB);
    unsigned short* wpk   = (unsigned short*)(ws + planeB + 2 * actB);
    const int perLayer = C * 9 * C;            // 147456 bf16 per packed layer

    // 1) pack conv2..conv6 weights to bf16 [oc][tap][ic]
    pack_weights_kernel<<<(5 * perLayer + 255) / 256, 256, 0, stream>>>(
        w2, w3, w4, w5, w6, wpk);

    // 2) stage-1 depth estimate (wave per pixel)
    stage1_kernel<<<(HW + 7) / 8, 256, 0, stream>>>(ma, mc, grid, angle, valid, plane);

    // 3) layer 1 (3->128, dil2)
    conv1_kernel<<<(int)(((long)HW * C + 255) / 256), 256, 0, stream>>>(
        plane, expd, img, w1, b1, actA);

    // 4) layers 2..6 (128->128) on the WMMA path, 64-px N tiles
    dim3 cgrid((W + 16 * NSUB - 1) / (16 * NSUB), H);
    conv_wmma_kernel<<<cgrid, 256, 0, stream>>>(actA,  actBf, wpk + 0 * perLayer, b2, 2);
    conv_wmma_kernel<<<cgrid, 256, 0, stream>>>(actBf, actA,  wpk + 1 * perLayer, b3, 2);
    conv_wmma_kernel<<<cgrid, 256, 0, stream>>>(actA,  actBf, wpk + 2 * perLayer, b4, 1);
    conv_wmma_kernel<<<cgrid, 256, 0, stream>>>(actBf, actA,  wpk + 3 * perLayer, b5, 1);
    conv_wmma_kernel<<<cgrid, 256, 0, stream>>>(actA,  actBf, wpk + 4 * perLayer, b6, 1);

    // 5) layer 7 (128->1), f32 output
    conv7_kernel<<<(HW + 255) / 256, 256, 0, stream>>>(actBf, w7, b7, out);
}
